// MultiHeadedRelativeAttention_24361054503480
// MI455X (gfx1250) — matrix operand order repaired and checked by reading, not compiled, not run on hardware
//
#include <hip/hip_runtime.h>
#include <hip/hip_bf16.h>
#include <stdint.h>

#define Bsz  2
#define Tseq 2048
#define Dm   1024
#define Hh   16
#define Dk   64
#define NR   17   // 2*8+1 relative-position buckets

typedef __attribute__((ext_vector_type(16))) __bf16         v16bf;
typedef __attribute__((ext_vector_type(8)))  float          v8f;
typedef __attribute__((ext_vector_type(16))) unsigned short v16us;

__device__ __forceinline__ unsigned short f2bf(float f) {
  return __builtin_bit_cast(unsigned short, (__bf16)f);   // native cvt, RNE
}
__device__ __forceinline__ float bf2f(unsigned short u) {
  return __uint_as_float((unsigned)u << 16);
}

// ---- WMMA operand loaders (contiguous k order, wave32) --------------------
// A (16x32 bf16): lane l -> row m=l%16; element e -> K = hi*16 + e
// => two contiguous 8-element (16B) chunks at k = hi*16 and k = hi*16+8.
__device__ __forceinline__ v16bf wload_A_bf(const unsigned short* row, int hi) {
  union { uint4 u[2]; v16us v; } t;
  t.u[0] = *(const uint4*)(row + hi * 16);
  t.u[1] = *(const uint4*)(row + hi * 16 + 8);
  return __builtin_bit_cast(v16bf, t.v);
}
__device__ __forceinline__ v16bf wload_A_f32(const float* row, int hi) {
  union { v16us v; unsigned short s[16]; } t;
#pragma unroll
  for (int i = 0; i < 8; ++i) {
    t.s[i]     = f2bf(row[hi * 16 + i]);
    t.s[8 + i] = f2bf(row[hi * 16 + 8 + i]);
  }
  return __builtin_bit_cast(v16bf, t.v);
}
// B (32x16 bf16): lane l -> col n=l%16; element e -> K = hi*16 + e
// => 16 contiguous bf16 (32B). Caller passes (colbase + hi*16).
__device__ __forceinline__ v16bf wload_B16(const unsigned short* p) {
  union { uint4 u[2]; v16us v; } t;
  t.u[0] = *(const uint4*)(p);
  t.u[1] = *(const uint4*)(p + 8);
  return __builtin_bit_cast(v16bf, t.v);
}
__device__ __forceinline__ v8f wmma_bf16(v16bf a, v16bf b, v8f c) {
  return __builtin_amdgcn_wmma_f32_16x16x32_bf16(false, a, false, b,
                                                 (short)0, c, false, false);
}

// ===========================================================================
// GEMM: C[4096,1024] = A[4096,1024] @ W[1024,1024] + bias
// MODE 0: Q   -> bf16 out [B,H,T,DK]
// MODE 1: K   -> bf16 out [B,H,T,DK]
// MODE 2: V   -> bf16 out [B,H,DK,T]  (transposed for PV B-operand)
// MODE 3: x(bf16) @ w_o  -> f32 out [B*T, Dm]
// block: 256 thr (8 waves), tile M=128 x N=64, K-step 32, double-buffered LDS.
// ===========================================================================
template <int MODE>
__global__ __launch_bounds__(256)
void proj_gemm(const float* __restrict__ A32,
               const unsigned short* __restrict__ A16,
               const float* __restrict__ W, const float* __restrict__ bias,
               void* __restrict__ out)
{
  __shared__ __align__(16) unsigned short s_wt[2][64 * 32]; // [buf][n][k] bf16
  const int tid  = threadIdx.x;
  const int lane = tid & 31, wv = tid >> 5;
  const int hi   = lane >> 4, ln = lane & 15;
  const int nbase = blockIdx.x * 64;
  const int mbase = blockIdx.y * 128;
  const int row   = mbase + wv * 16 + ln;    // A-operand row for this lane
  const int sn  = tid & 63;                  // staging: this thread's column
  const int sk0 = (tid >> 6) * 8;            // staging: this thread's k-group

  // stage W[kk:kk+32, nbase:nbase+64] transposed -> s_wt[buf][n][k]:
  // 8 coalesced b32 loads (stride Dm), pack 8 bf16, one ds_store_b128.
  auto stage = [&](int buf, int kk) {
    union { uint4 u; unsigned short s[8]; } t;
    const float* wp = W + (size_t)(kk + sk0) * Dm + nbase + sn;
#pragma unroll
    for (int i = 0; i < 8; ++i) t.s[i] = f2bf(wp[(size_t)i * Dm]);
    *(uint4*)&s_wt[buf][sn * 32 + sk0] = t.u;
  };

  v8f acc[4] = {};
  stage(0, 0);
  int pb = 0;
  for (int kk = 0; kk < Dm; kk += 32) {
    __syncthreads();                          // staged tile visible
    if (kk + 32 < Dm) stage(pb ^ 1, kk + 32); // prefetch next under WMMAs

    v16bf a;
    if (MODE < 3) a = wload_A_f32(A32 + (size_t)row * Dm + kk, hi);
    else          a = wload_A_bf (A16 + (size_t)row * Dm + kk, hi);

#pragma unroll
    for (int nt = 0; nt < 4; ++nt) {
      v16bf bw = wload_B16(&s_wt[pb][(nt * 16 + ln) * 32 + hi * 16]);
      acc[nt] = wmma_bf16(a, bw, acc[nt]);
    }
    pb ^= 1;
  }

  // epilogue: C layout -> row m = r + 8*hi, col n = ln (per 16-col tile)
#pragma unroll
  for (int nt = 0; nt < 4; ++nt) {
    int ng = nbase + nt * 16 + ln;
    float bv = bias[ng];
#pragma unroll
    for (int r = 0; r < 8; ++r) {
      int rg = mbase + wv * 16 + r + hi * 8;
      float val = acc[nt][r] + bv;
      if constexpr (MODE == 3) {
        ((float*)out)[(size_t)rg * Dm + ng] = val;
      } else {
        int b = rg >> 11, t = rg & (Tseq - 1);
        int h = ng >> 6,  dk = ng & 63;
        size_t bh = (size_t)(b * Hh + h);
        unsigned short v = f2bf(val);
        if constexpr (MODE == 2)
          ((unsigned short*)out)[(bh * Dk + dk) * Tseq + t] = v;
        else
          ((unsigned short*)out)[(bh * Tseq + t) * Dk + dk] = v;
      }
    }
  }
}

// ===========================================================================
// Fused flash attention with Shaw relative-position bias.
// grid (T/64, H, B), block 128 (4 waves, each owns 16 queries).
// Diagonal-band specialization: only key tiles intersecting
// s in [q0-8, q0+23] need per-element buckets; everywhere else the bucket is
// uniformly 0 (left) or 16 (right), handled with register-held constants.
// ===========================================================================
__global__ __launch_bounds__(128)
void flash_rpr(const unsigned short* __restrict__ Q,   // [B,H,T,DK] bf16
               const unsigned short* __restrict__ K,   // [B,H,T,DK] bf16
               const unsigned short* __restrict__ Vt,  // [B,H,DK,T] bf16
               const unsigned char*  __restrict__ mask,// [B,T] bool
               const float* __restrict__ tabK,         // [17,64]
               const float* __restrict__ tabV,         // [17,64]
               unsigned short* __restrict__ Xo)        // [B,T,Dm] bf16
{
  __shared__ __align__(16) float s_tk[NR * Dk];
  __shared__ __align__(16) float s_tv[NR * Dk];
  __shared__ __align__(16) float s_rq[4][16 * NR];       // q . ek[r]
  __shared__ __align__(16) float s_w [4][16 * NR];       // diag bucket mass
  __shared__ __align__(16) unsigned short s_p[4][16 * 64]; // P transpose tile

  const int tid  = threadIdx.x;
  const int lane = tid & 31, wv = tid >> 5;
  const int hi   = lane >> 4, ln = lane & 15;
  const int h = blockIdx.y, b = blockIdx.z;
  const int bh = b * Hh + h;
  const int q0 = blockIdx.x * 64 + wv * 16;

  const unsigned short* Qp = Q  + (size_t)bh * Tseq * Dk;
  const unsigned short* Kp = K  + (size_t)bh * Tseq * Dk;
  const unsigned short* Vp = Vt + (size_t)bh * Dk * Tseq;
  const unsigned char*  mp = mask + (size_t)b * Tseq;

  for (int i = tid; i < NR * Dk; i += 128) { s_tk[i] = tabK[i]; s_tv[i] = tabV[i]; }
  __syncthreads();

  // rq[m][r] = q[q0+m,:] . tabK[r,:]   (272 dots of length 64 per wave)
  for (int e = lane; e < 16 * NR; e += 32) {
    int m = e / NR, r = e % NR;
    const unsigned short* qr = Qp + (size_t)(q0 + m) * Dk;
    float acc = 0.f;
#pragma unroll 8
    for (int k = 0; k < Dk; ++k) acc += bf2f(qr[k]) * s_tk[r * Dk + k];
    s_rq[wv][e] = acc;
    s_w[wv][e]  = 0.f;
  }

  // Q A-operands held in registers for the whole key loop
  const unsigned short* qrow = Qp + (size_t)(q0 + ln) * Dk;
  v16bf aq0 = wload_A_bf(qrow, hi);
  v16bf aq1 = wload_A_bf(qrow + 32, hi);

  // per-row edge biases (bucket 0 / bucket 16), register-resident
  float rqlo[8], rqhi[8];
#pragma unroll
  for (int r = 0; r < 8; ++r) {
    int m = r + hi * 8;
    rqlo[r] = s_rq[wv][m * NR + 0];
    rqhi[r] = s_rq[wv][m * NR + 16];
  }

  float rmax[8], lsum[8], wlo[8], whi[8];
#pragma unroll
  for (int r = 0; r < 8; ++r) { rmax[r] = -1e30f; lsum[r] = 0.f; wlo[r] = 0.f; whi[r] = 0.f; }
  v8f o[4] = {};
  bool wseen = false;   // any diagonal-tile mass deposited into s_w yet?

  for (int kb = 0; kb < Tseq; kb += 64) {
    const bool diag  = (kb + 71 > q0) && (kb < q0 + 23); // intersects band
    const bool right = kb > q0;                          // fast tile side

    if (kb + 64 < Tseq) {
      __builtin_prefetch(Kp + (size_t)(kb + 64 + lane) * Dk, 0, 0);
      __builtin_prefetch(Vp + (size_t)lane * Tseq + kb + 64, 0, 0);
    }

    // ---- S = Q K^T (per 16-key sub-tile), + rel bias, scale, mask ----
    v8f s[4];
#pragma unroll
    for (int j = 0; j < 4; ++j) {
      int sk = kb + j * 16 + ln;
      const unsigned short* kr = Kp + (size_t)sk * Dk;   // dk-contiguous
      v8f c = {};
      c = wmma_bf16(aq0, wload_B16(kr + hi * 16), c);
      c = wmma_bf16(aq1, wload_B16(kr + 32 + hi * 16), c);
      bool mok = mp[sk] != 0;
      if (diag) {
#pragma unroll
        for (int r = 0; r < 8; ++r) {
          int m = r + hi * 8;
          int d = sk - (q0 + m) + 8;
          d = d < 0 ? 0 : (d > 16 ? 16 : d);
          float val = (c[r] + s_rq[wv][m * NR + d]) * 0.125f;
          c[r] = mok ? val : -1e9f;
        }
      } else {
#pragma unroll
        for (int r = 0; r < 8; ++r) {
          float val = (c[r] + (right ? rqhi[r] : rqlo[r])) * 0.125f;
          c[r] = mok ? val : -1e9f;
        }
      }
      s[j] = c;
    }

    // ---- online softmax: row max across 16 lanes of the half ----
    float tmax[8];
#pragma unroll
    for (int r = 0; r < 8; ++r)
      tmax[r] = fmaxf(fmaxf(s[0][r], s[1][r]), fmaxf(s[2][r], s[3][r]));
#pragma unroll
    for (int msk = 1; msk <= 8; msk <<= 1)
#pragma unroll
      for (int r = 0; r < 8; ++r)
        tmax[r] = fmaxf(tmax[r], __shfl_xor(tmax[r], msk, 32));

    float scl[8];
#pragma unroll
    for (int r = 0; r < 8; ++r) {
      float mn = fmaxf(rmax[r], tmax[r]);
      scl[r] = __expf(rmax[r] - mn);
      rmax[r] = mn;
      lsum[r] *= scl[r];
      wlo[r]  *= scl[r];
      whi[r]  *= scl[r];
#pragma unroll
      for (int nt = 0; nt < 4; ++nt) o[nt][r] *= scl[r];
    }
    if (wseen) {   // rescale diag bucket masses (half hi owns rows hi*8..+7)
      for (int e = ln; e < 8 * NR; e += 16) {
        int rl = e / NR;
        s_w[wv][(hi * 8 + rl) * NR + (e % NR)] *= scl[rl];
      }
    }

    // ---- P = exp(S-m): row sums, bucket accumulation, LDS transpose ----
    float psum[8] = {0, 0, 0, 0, 0, 0, 0, 0};
    if (diag) {
#pragma unroll
      for (int j = 0; j < 4; ++j) {
        int sk = kb + j * 16 + ln;
#pragma unroll
        for (int r = 0; r < 8; ++r) {
          int m = r + hi * 8;
          float p = __expf(s[j][r] - rmax[r]);
          psum[r] += p;
          int d = sk - (q0 + m) + 8;
          d = d < 0 ? 0 : (d > 16 ? 16 : d);
          atomicAdd(&s_w[wv][m * NR + d], p);          // ds_add_f32
          s_p[wv][m * 64 + j * 16 + ln] = f2bf(p);
        }
      }
      wseen = true;
    } else {
#pragma unroll
      for (int j = 0; j < 4; ++j) {
#pragma unroll
        for (int r = 0; r < 8; ++r) {
          int m = r + hi * 8;
          float p = __expf(s[j][r] - rmax[r]);
          psum[r] += p;
          s_p[wv][m * 64 + j * 16 + ln] = f2bf(p);
        }
      }
    }
#pragma unroll
    for (int msk = 1; msk <= 8; msk <<= 1)
#pragma unroll
      for (int r = 0; r < 8; ++r)
        psum[r] += __shfl_xor(psum[r], msk, 32);
#pragma unroll
    for (int r = 0; r < 8; ++r) {
      lsum[r] += psum[r];
      if (!diag) { if (right) whi[r] += psum[r]; else wlo[r] += psum[r]; }
    }

    asm volatile("s_wait_dscnt 0" ::: "memory");     // P tile in LDS complete

    // ---- O += P @ V (contraction over 64 keys, Vt is s-contiguous) ----
    const unsigned short* pr = &s_p[wv][ln * 64];
    v16bf ap0 = wload_A_bf(pr, hi);
    v16bf ap1 = wload_A_bf(pr + 32, hi);
#pragma unroll
    for (int nt = 0; nt < 4; ++nt) {
      const unsigned short* vr = Vp + (size_t)(nt * 16 + ln) * Tseq + kb;
      o[nt] = wmma_bf16(ap0, wload_B16(vr + hi * 16), o[nt]);
      o[nt] = wmma_bf16(ap1, wload_B16(vr + 32 + hi * 16), o[nt]);
    }
  }

  asm volatile("s_wait_dscnt 0" ::: "memory");

  // ---- epilogue: O + (bucket masses @ ev_table), normalize, store bf16 x ---
#pragma unroll
  for (int nt = 0; nt < 4; ++nt) {
#pragma unroll
    for (int r = 0; r < 8; ++r) {
      int m = r + hi * 8;
      int n = nt * 16 + ln;
      float acc = o[nt][r]
                + wlo[r] * s_tv[0 * Dk + n]
                + whi[r] * s_tv[16 * Dk + n];
#pragma unroll
      for (int rr = 0; rr < NR; ++rr)
        acc += s_w[wv][m * NR + rr] * s_tv[rr * Dk + n];
      acc /= lsum[r];
      Xo[(size_t)(b * Tseq + q0 + m) * Dm + h * Dk + n] = f2bf(acc);
    }
  }
}

// ===========================================================================
extern "C" void kernel_launch(void* const* d_in, const int* in_sizes, int n_in,
                              void* d_out, int out_size, void* d_ws, size_t ws_size,
                              hipStream_t stream) {
  const float* query = (const float*)d_in[0];
  const float* key   = (const float*)d_in[1];
  const float* value = (const float*)d_in[2];
  const unsigned char* mask = (const unsigned char*)d_in[3];
  const float* w_q  = (const float*)d_in[4];
  const float* b_q  = (const float*)d_in[5];
  const float* w_k  = (const float*)d_in[6];
  const float* b_k  = (const float*)d_in[7];
  const float* w_v  = (const float*)d_in[8];
  const float* b_v  = (const float*)d_in[9];
  const float* w_o  = (const float*)d_in[10];
  const float* b_o  = (const float*)d_in[11];
  const float* tabK = (const float*)d_in[12];
  const float* tabV = (const float*)d_in[13];

  // workspace carve-up (each region 8 MB, 32 MB total)
  char* ws = (char*)d_ws;
  unsigned short* q_ws  = (unsigned short*)(ws);
  unsigned short* k_ws  = (unsigned short*)(ws + (size_t)(8u  << 20));
  unsigned short* vt_ws = (unsigned short*)(ws + (size_t)(16u << 20));
  unsigned short* x_ws  = (unsigned short*)(ws + (size_t)(24u << 20));

  dim3 gg(Dm / 64, (Bsz * Tseq) / 128);
  proj_gemm<0><<<gg, 256, 0, stream>>>(query, nullptr, w_q, b_q, q_ws);
  proj_gemm<1><<<gg, 256, 0, stream>>>(key,   nullptr, w_k, b_k, k_ws);
  proj_gemm<2><<<gg, 256, 0, stream>>>(value, nullptr, w_v, b_v, vt_ws);

  flash_rpr<<<dim3(Tseq / 64, Hh, Bsz), 128, 0, stream>>>(
      q_ws, k_ws, vt_ws, mask, tabK, tabV, x_ws);

  proj_gemm<3><<<gg, 256, 0, stream>>>(nullptr, x_ws, w_o, b_o, d_out);
}
